// ManualTransformerLayer_86242943303814
// MI455X (gfx1250) — compile-verified, matrix operand order checked by reading, not run on hardware
//
#include <hip/hip_runtime.h>
#include <hip/hip_bf16.h>

// ---------------------------------------------------------------------------
// ManualTransformerLayer for MI455X (gfx1250, wave32, WMMA + async-to-LDS)
// E=1024, F=4096, H=16, Dh=64, tokens NT = 2*2048 = 4096
// ---------------------------------------------------------------------------

#define E_DIM 1024
#define F_DIM 4096
#define NT    4096
#define H_NUM 16
#define DH    64
#define L_SEQ 2048

typedef __attribute__((ext_vector_type(16))) _Float16 v16h;
typedef __attribute__((ext_vector_type(8)))  float    v8f;

union Frag {
    v16h v;
    unsigned u[8];
    _Float16 h[16];
    __device__ Frag() {}
};

// CDNA5 async copy: 16B global -> LDS, tracked by ASYNCcnt (no VGPR round-trip)
__device__ __forceinline__ void async_copy_b128(unsigned lds_off, const void* gaddr) {
    asm volatile("global_load_async_to_lds_b128 %0, %1, off"
                 :: "v"(lds_off), "v"((unsigned long long)gaddr)
                 : "memory");
}
__device__ __forceinline__ void wait_async0() {
    asm volatile("s_wait_asynccnt 0x0" ::: "memory");
}
__device__ __forceinline__ unsigned lds_addr(const void* p) {
    return (unsigned)(unsigned long long)p;   // low 32 bits = LDS offset
}

// --------------------------- f32 -> f16 convert ----------------------------
__global__ void cvt_f32_f16(const float* __restrict__ in,
                            _Float16* __restrict__ out, int n) {
    int i = blockIdx.x * blockDim.x + threadIdx.x;
    if (i < n) out[i] = (_Float16)in[i];
}

// f32 [R][C] -> f16 transposed [C][R] (one-time; amortized over 32 A-tiles)
__global__ void cvt_t_f32_f16(const float* __restrict__ in,
                              _Float16* __restrict__ out, int R, int C) {
    int i = blockIdx.x * blockDim.x + threadIdx.x;
    if (i < R * C) {
        int r = i / C, c = i % C;
        out[(size_t)c * R + r] = (_Float16)in[i];
    }
}

// --------------------------- WMMA tiled GEMM -------------------------------
// C[M,N] = A[M,K] @ Bt^T + bias, A f16 row-major [M][K], Bt f16 [N][K].
// mode 0: f32 out; mode 1: f16 out; mode 2: relu + f16 out.
// Block: 256 threads = 8 waves; tile 128(M) x 64(N); wave sub-tile 32x32.
__global__ __launch_bounds__(256) void gemm_f16(
    const _Float16* __restrict__ A, const _Float16* __restrict__ Bt,
    const float* __restrict__ bias, float* __restrict__ outF,
    _Float16* __restrict__ outH, int M, int N, int K, int mode)
{
    __shared__ _Float16 As[128 * 32];  // [m][k]
    __shared__ _Float16 Bs[64 * 32];   // [n][k]  (B^T layout)

    const int tid  = threadIdx.x;
    const int lane = tid & 31;
    const int wave = tid >> 5;
    const int hf   = lane >> 4;
    const int ml   = lane & 15;
    const int kb   = hf * 8;
    const int wm   = (wave >> 1) * 32;   // wave rows:  0/32/64/96
    const int wn   = (wave & 1) * 32;    // wave cols:  0/32
    const int gm0  = blockIdx.y * 128;
    const int gn0  = blockIdx.x * 64;

    // async-copy chunk assignments (16B chunks)
    const int ar0 = tid >> 1,  aseg0 = (tid & 1) * 8;           // A chunks 0..255
    const int ar1 = (tid + 256) >> 1, aseg1 = ((tid + 256) & 1) * 8; // unused form
    const int br  = tid >> 2,  bseg  = (tid & 3) * 8;           // B chunks 0..255

    v8f acc[2][2];
    const v8f vzero = {0.f, 0.f, 0.f, 0.f, 0.f, 0.f, 0.f, 0.f};
#pragma unroll
    for (int i = 0; i < 2; ++i)
#pragma unroll
        for (int j = 0; j < 2; ++j) acc[i][j] = vzero;

    for (int k0 = 0; k0 < K; k0 += 32) {
        __syncthreads();   // previous iteration's fragment reads complete
        // A tile 128x32: 512 chunks of 16B, 2 per thread
        {
            const int c0 = tid, c1 = tid + 256;
            const int r0 = c0 >> 2, s0 = (c0 & 3) * 8;
            const int r1 = c1 >> 2, s1 = (c1 & 3) * 8;
            async_copy_b128(lds_addr(&As[r0 * 32 + s0]),
                            &A[(size_t)(gm0 + r0) * K + k0 + s0]);
            async_copy_b128(lds_addr(&As[r1 * 32 + s1]),
                            &A[(size_t)(gm0 + r1) * K + k0 + s1]);
        }
        // B tile 64x32 ([n][k]): 256 chunks of 16B, 1 per thread
        async_copy_b128(lds_addr(&Bs[br * 32 + bseg]),
                        &Bt[(size_t)(gn0 + br) * K + k0 + bseg]);
        wait_async0();
        __syncthreads();   // all waves' async tiles visible

        Frag a[2], b[2];
#pragma unroll
        for (int im = 0; im < 2; ++im)
#pragma unroll
            for (int j = 0; j < 4; ++j) {
                const int row = wm + im * 16 + ml;
                a[im].u[j]     = *(const unsigned*)&As[row * 32 + kb + 2 * j];
                a[im].u[4 + j] = *(const unsigned*)&As[row * 32 + kb + 16 + 2 * j];
            }
#pragma unroll
        for (int in = 0; in < 2; ++in)
#pragma unroll
            for (int j = 0; j < 4; ++j) {
                const int col = wn + in * 16 + ml;
                b[in].u[j]     = *(const unsigned*)&Bs[col * 32 + kb + 2 * j];
                b[in].u[4 + j] = *(const unsigned*)&Bs[col * 32 + kb + 16 + 2 * j];
            }
#pragma unroll
        for (int im = 0; im < 2; ++im)
#pragma unroll
            for (int in = 0; in < 2; ++in)
                acc[im][in] = __builtin_amdgcn_wmma_f32_16x16x32_f16(
                    false, a[im].v, false, b[in].v, (short)0, acc[im][in],
                    false, false);
    }
    (void)ar0; (void)aseg0; (void)ar1; (void)aseg1;

#pragma unroll
    for (int im = 0; im < 2; ++im)
#pragma unroll
        for (int in = 0; in < 2; ++in) {
            const int colg = gn0 + wn + in * 16 + ml;
            const float bcol = bias ? bias[colg] : 0.0f;
#pragma unroll
            for (int i = 0; i < 8; ++i) {
                const int rowg = gm0 + wm + im * 16 + i + 8 * hf;
                const float val = acc[im][in][i] + bcol;
                const size_t idx = (size_t)rowg * N + colg;
                if (mode == 0)      outF[idx] = val;
                else if (mode == 1) outH[idx] = (_Float16)val;
                else                outH[idx] = (_Float16)fmaxf(val, 0.0f);
            }
        }
}

// --------------------------- flash attention -------------------------------
// scores = Q @ V^T / 8 (reference quirk: V, not K), softmax, ctx = P @ V.
// grid = (L/64, B*H); block = 128 threads = 4 waves; each wave owns 16 q-rows.
__global__ __launch_bounds__(128) void attn_kernel(
    const _Float16* __restrict__ Q, const _Float16* __restrict__ V,
    _Float16* __restrict__ O)
{
    __shared__ _Float16 Vlds[64 * 64];       // [key][dh]
    __shared__ _Float16 Plds[4 * 16 * 64];   // per-wave [row][key]

    const int tid  = threadIdx.x;
    const int lane = tid & 31;
    const int wave = tid >> 5;
    const int hf   = lane >> 4;
    const int ml   = lane & 15;
    const int kb   = hf * 8;

    const int bh    = blockIdx.y;         // b*16 + h
    const int bb    = bh >> 4;
    const int hh    = bh & 15;
    const int qtok0 = bb * L_SEQ + blockIdx.x * 64;
    const int vtok0 = bb * L_SEQ;
    const size_t hoff = (size_t)hh * DH;

    // Q fragments: wave's 16 rows x Dh=64 (two K-steps of 32)
    Frag aq[2];
    {
        const size_t qrow = (size_t)(qtok0 + wave * 16 + ml) * E_DIM + hoff;
#pragma unroll
        for (int ks = 0; ks < 2; ++ks)
#pragma unroll
            for (int j = 0; j < 4; ++j) {
                aq[ks].u[j]     = *(const unsigned*)&Q[qrow + ks * 32 + kb + 2 * j];
                aq[ks].u[4 + j] = *(const unsigned*)&Q[qrow + ks * 32 + kb + 16 + 2 * j];
            }
    }

    v8f acc[4];
    const v8f vzero = {0.f, 0.f, 0.f, 0.f, 0.f, 0.f, 0.f, 0.f};
#pragma unroll
    for (int t = 0; t < 4; ++t) acc[t] = vzero;
    float mi[8], li[8];
#pragma unroll
    for (int i = 0; i < 8; ++i) { mi[i] = -3.0e38f; li[i] = 0.0f; }

    for (int kc = 0; kc < L_SEQ / 64; ++kc) {
        // async-load V chunk [64 keys][64 dh]: 512 x 16B chunks, 4 per thread
#pragma unroll
        for (int j = 0; j < 4; ++j) {
            const int c = tid + j * 128;
            const int r = c >> 3, seg = (c & 7) * 8;
            async_copy_b128(lds_addr(&Vlds[r * 64 + seg]),
                            &V[(size_t)(vtok0 + kc * 64 + r) * E_DIM + hoff + seg]);
        }
        wait_async0();
        __syncthreads();

        // S = Q @ Vchunk^T  (Vlds row-major IS the B^T layout here)
        v8f s[4];
#pragma unroll
        for (int t = 0; t < 4; ++t) {
            v8f sv = vzero;
            const int key = t * 16 + ml;
#pragma unroll
            for (int ks = 0; ks < 2; ++ks) {
                Frag bf;
#pragma unroll
                for (int j = 0; j < 4; ++j) {
                    bf.u[j]     = *(const unsigned*)&Vlds[key * 64 + ks * 32 + kb + 2 * j];
                    bf.u[4 + j] = *(const unsigned*)&Vlds[key * 64 + ks * 32 + kb + 16 + 2 * j];
                }
                sv = __builtin_amdgcn_wmma_f32_16x16x32_f16(
                    false, aq[ks].v, false, bf.v, (short)0, sv, false, false);
            }
            s[t] = sv;
        }
        // scale by 1/sqrt(Dh)
#pragma unroll
        for (int t = 0; t < 4; ++t)
#pragma unroll
            for (int i = 0; i < 8; ++i) s[t][i] *= 0.125f;

        // online softmax per row (rows live per-vgpr; reduce across 16-lane half)
#pragma unroll
        for (int i = 0; i < 8; ++i) {
            float cm = fmaxf(fmaxf(s[0][i], s[1][i]), fmaxf(s[2][i], s[3][i]));
            for (int off = 1; off < 16; off <<= 1)
                cm = fmaxf(cm, __shfl_xor(cm, off, 32));
            const float mn = fmaxf(mi[i], cm);
            const float sc = __expf(mi[i] - mn);
            float rs = 0.0f;
#pragma unroll
            for (int t = 0; t < 4; ++t) {
                const float p = __expf(s[t][i] - mn);
                s[t][i] = p;
                rs += p;
            }
            for (int off = 1; off < 16; off <<= 1)
                rs += __shfl_xor(rs, off, 32);
            li[i] = li[i] * sc + rs;
            mi[i] = mn;
#pragma unroll
            for (int t = 0; t < 4; ++t) acc[t][i] *= sc;
        }

        // re-layout P: C-layout -> LDS -> A-layout (wave-local, no barrier)
        _Float16* pw = &Plds[wave * 16 * 64];
#pragma unroll
        for (int t = 0; t < 4; ++t)
#pragma unroll
            for (int i = 0; i < 8; ++i)
                pw[(i + 8 * hf) * 64 + t * 16 + ml] = (_Float16)s[t][i];

        Frag ap[2];
#pragma unroll
        for (int ks = 0; ks < 2; ++ks)
#pragma unroll
            for (int j = 0; j < 4; ++j) {
                ap[ks].u[j]     = *(const unsigned*)&pw[ml * 64 + ks * 32 + kb + 2 * j];
                ap[ks].u[4 + j] = *(const unsigned*)&pw[ml * 64 + ks * 32 + kb + 16 + 2 * j];
            }

        // ctx += P @ Vchunk   (B in natural [K][N] layout -> strided gathers)
#pragma unroll
        for (int t = 0; t < 4; ++t) {
            const int dhc = t * 16 + ml;
#pragma unroll
            for (int ks = 0; ks < 2; ++ks) {
                Frag bf;
#pragma unroll
                for (int f = 0; f < 8; ++f) {
                    bf.h[f]     = Vlds[(ks * 32 + kb + f) * 64 + dhc];
                    bf.h[f + 8] = Vlds[(ks * 32 + kb + 16 + f) * 64 + dhc];
                }
                acc[t] = __builtin_amdgcn_wmma_f32_16x16x32_f16(
                    false, ap[ks].v, false, bf.v, (short)0, acc[t], false, false);
            }
        }
        __syncthreads();
    }

    // normalize and store ctx (f16, [token, h*64+dh] layout)
#pragma unroll
    for (int t = 0; t < 4; ++t)
#pragma unroll
        for (int i = 0; i < 8; ++i) {
            const int tok = qtok0 + wave * 16 + i + 8 * hf;
            O[(size_t)tok * E_DIM + hoff + t * 16 + ml] =
                (_Float16)(acc[t][i] / li[i]);
        }
}

// ----------------------- residual + LayerNorm ------------------------------
__global__ __launch_bounds__(256) void res_ln_kernel(
    const float* __restrict__ resid, const float* __restrict__ y,
    const float* __restrict__ g, const float* __restrict__ be,
    float* __restrict__ outF, _Float16* __restrict__ outH)
{
    __shared__ float r1[256], r2[256];
    const int row = blockIdx.x, tid = threadIdx.x;
    float v[4];
    float sa = 0.0f, sb = 0.0f;
#pragma unroll
    for (int j = 0; j < 4; ++j) {
        const int c = tid + j * 256;
        const float t = resid[(size_t)row * E_DIM + c] + y[(size_t)row * E_DIM + c];
        v[j] = t; sa += t; sb += t * t;
    }
    r1[tid] = sa; r2[tid] = sb;
    __syncthreads();
    for (int off = 128; off > 0; off >>= 1) {
        if (tid < off) { r1[tid] += r1[tid + off]; r2[tid] += r2[tid + off]; }
        __syncthreads();
    }
    const float mu  = r1[0] * (1.0f / E_DIM);
    const float var = r2[0] * (1.0f / E_DIM) - mu * mu;
    const float rs  = rsqrtf(var + 1e-5f);
#pragma unroll
    for (int j = 0; j < 4; ++j) {
        const int c = tid + j * 256;
        const float o = (v[j] - mu) * rs * g[c] + be[c];
        outF[(size_t)row * E_DIM + c] = o;
        if (outH) outH[(size_t)row * E_DIM + c] = (_Float16)o;
    }
}

// ---------------------------------------------------------------------------
extern "C" void kernel_launch(void* const* d_in, const int* in_sizes, int n_in,
                              void* d_out, int out_size, void* d_ws, size_t ws_size,
                              hipStream_t stream)
{
    (void)in_sizes; (void)n_in; (void)out_size; (void)ws_size;
    const float* x   = (const float*)d_in[0];
    const float* Wq  = (const float*)d_in[1];
    const float* bq  = (const float*)d_in[2];
    // d_in[3]=Wk, d_in[4]=bk: computed-but-unused in reference -> skipped
    const float* Wv  = (const float*)d_in[5];
    const float* bv  = (const float*)d_in[6];
    const float* Wo  = (const float*)d_in[7];
    const float* bo  = (const float*)d_in[8];
    const float* g1  = (const float*)d_in[9];
    const float* b1  = (const float*)d_in[10];
    const float* W1  = (const float*)d_in[11];
    const float* bf1 = (const float*)d_in[12];
    const float* W2  = (const float*)d_in[13];
    const float* bf2 = (const float*)d_in[14];
    const float* g2  = (const float*)d_in[15];
    const float* b2  = (const float*)d_in[16];
    float* out = (float*)d_out;

    char* ws = (char*)d_ws;
    const size_t MB = 1024u * 1024u;
    _Float16* xb   = (_Float16*)(ws + 0 * MB);    // 8 MB  [NT][E]
    _Float16* Wqt  = (_Float16*)(ws + 8 * MB);    // 2 MB  [E][E]  (transposed)
    _Float16* Wvt  = (_Float16*)(ws + 10 * MB);   // 2 MB
    _Float16* Wot  = (_Float16*)(ws + 12 * MB);   // 2 MB
    _Float16* W1t  = (_Float16*)(ws + 14 * MB);   // 8 MB  [F][E]
    _Float16* W2t  = (_Float16*)(ws + 22 * MB);   // 8 MB  [E][F]
    _Float16* qb   = (_Float16*)(ws + 30 * MB);   // 8 MB
    _Float16* vbh  = (_Float16*)(ws + 38 * MB);   // 8 MB
    _Float16* ctxb = (_Float16*)(ws + 46 * MB);   // 8 MB
    float*    tmp  = (float*)   (ws + 54 * MB);   // 16 MB
    float*    pa   = (float*)   (ws + 70 * MB);   // 16 MB
    _Float16* pah  = (_Float16*)(ws + 86 * MB);   // 8 MB
    _Float16* ff1  = (_Float16*)(ws + 94 * MB);   // 32 MB  (total 126 MB)

    // precision conversion (+ one-time weight transpose to [N][K])
    cvt_f32_f16<<<(NT * E_DIM) / 256, 256, 0, stream>>>(x, xb, NT * E_DIM);
    cvt_t_f32_f16<<<(E_DIM * E_DIM) / 256, 256, 0, stream>>>(Wq, Wqt, E_DIM, E_DIM);
    cvt_t_f32_f16<<<(E_DIM * E_DIM) / 256, 256, 0, stream>>>(Wv, Wvt, E_DIM, E_DIM);
    cvt_t_f32_f16<<<(E_DIM * E_DIM) / 256, 256, 0, stream>>>(Wo, Wot, E_DIM, E_DIM);
    cvt_t_f32_f16<<<(E_DIM * F_DIM) / 256, 256, 0, stream>>>(W1, W1t, E_DIM, F_DIM);
    cvt_t_f32_f16<<<(F_DIM * E_DIM) / 256, 256, 0, stream>>>(W2, W2t, F_DIM, E_DIM);

    // q = x@Wq + bq ; v = x@Wv + bv   (f16 outputs for attention WMMA)
    gemm_f16<<<dim3(E_DIM / 64, NT / 128), 256, 0, stream>>>(
        xb, Wqt, bq, nullptr, qb, NT, E_DIM, E_DIM, 1);
    gemm_f16<<<dim3(E_DIM / 64, NT / 128), 256, 0, stream>>>(
        xb, Wvt, bv, nullptr, vbh, NT, E_DIM, E_DIM, 1);

    // attention (scores = Q@V^T per reference)
    attn_kernel<<<dim3(L_SEQ / 64, 2 * H_NUM), 128, 0, stream>>>(qb, vbh, ctxb);

    // att_out = ctx@Wo + bo (f32), then post_att = LN(x + att_out)
    gemm_f16<<<dim3(E_DIM / 64, NT / 128), 256, 0, stream>>>(
        ctxb, Wot, bo, tmp, nullptr, NT, E_DIM, E_DIM, 0);
    res_ln_kernel<<<NT, 256, 0, stream>>>(x, tmp, g1, b1, pa, pah);

    // ff = relu(pa@W1 + bf1) @ W2 + bf2 ; out = LN(pa + ff)
    gemm_f16<<<dim3(F_DIM / 64, NT / 128), 256, 0, stream>>>(
        pah, W1t, bf1, nullptr, ff1, NT, F_DIM, E_DIM, 2);
    gemm_f16<<<dim3(E_DIM / 64, NT / 128), 256, 0, stream>>>(
        ff1, W2t, bf2, tmp, nullptr, NT, E_DIM, F_DIM, 0);
    res_ln_kernel<<<NT, 256, 0, stream>>>(pa, tmp, g2, b2, out, nullptr);
}
